// RoutedExperts_62380105007767
// MI455X (gfx1250) — matrix operand order repaired and checked by reading, not compiled, not run on hardware
//
#include <hip/hip_runtime.h>
#include <hip/hip_bf16.h>

// MoE expert MLP on gfx1250 (MI455X):
//   K1: up/gate GEMM + SiLU fused -> bf16 hidden in d_ws
//   K2: down GEMM (A tile staged by the Tensor Data Mover, tensor_load_to_lds)
// fp32 inputs converted to bf16 while staging; f32 accumulation via
// v_wmma_f32_16x16x32_bf16. Double-buffered LDS, one barrier per K-step.

typedef __bf16 bf16_t;
typedef __attribute__((ext_vector_type(16))) __bf16 v16bf;
typedef __attribute__((ext_vector_type(8)))  __bf16 v8bf;
typedef __attribute__((ext_vector_type(4)))  __bf16 v4bf;
typedef __attribute__((ext_vector_type(8)))  float  v8f;
typedef __attribute__((ext_vector_type(4)))  float  v4f;
typedef __attribute__((ext_vector_type(4)))  unsigned int u32x4;
typedef __attribute__((ext_vector_type(8)))  unsigned int u32x8;
typedef __attribute__((address_space(3)))    void lds_void;

#define N_TOK 32768
#define DDIM  2048
#define HDIM  4096
#define NEXP  8
#define TPE   (N_TOK / NEXP)   // 4096 tokens per expert

// LDS tile row stride in bf16 elements: 32 data + 8 pad = 80 B (16B-aligned rows)
#define LSTR 40

union FragBF {
    v16bf v;
    struct { v8bf lo, hi; } h;
};

__device__ __forceinline__ float silu_f(float g) {
    float e = __builtin_amdgcn_exp2f(-g * 1.44269504088896340736f);
    return g / (1.0f + e);
}

__device__ __forceinline__ v8f wmma_bf16(const FragBF& a, const FragBF& b, v8f c) {
    return __builtin_amdgcn_wmma_f32_16x16x32_bf16(
        false, a.v, false, b.v, (short)0, c, false, false);
}

// ---------------------------------------------------------------------------
// Kernel 1: up_gate = X_e * W_ug_e^T ; hidden = up * silu(gate) (bf16 out)
// Block: 128 tokens x 64 h (both up and gate halves). 8 waves (4x2).
// ---------------------------------------------------------------------------
__device__ __forceinline__ void stage_k1(const float* __restrict__ Ag,
                                         const float* __restrict__ Wg,
                                         int n0, int k0, int tid,
                                         bf16_t (*dA)[LSTR], bf16_t (*dB)[LSTR]) {
    const int ar = tid >> 3;        // 0..31
    const int ac = (tid & 7) * 4;   // 0..28
    #pragma unroll
    for (int i = 0; i < 4; ++i) {
        int r = ar + i * 32;
        v4f v = *(const v4f*)&Ag[(size_t)r * DDIM + k0 + ac];
        v4bf b;
        b[0] = (bf16_t)v[0]; b[1] = (bf16_t)v[1];
        b[2] = (bf16_t)v[2]; b[3] = (bf16_t)v[3];
        *(v4bf*)&dA[r][ac] = b;
    }
    #pragma unroll
    for (int i = 0; i < 4; ++i) {
        int r = ar + i * 32;
        int hrow = (r < 64) ? (n0 + r) : (HDIM + n0 + (r - 64));
        v4f v = *(const v4f*)&Wg[(size_t)hrow * DDIM + k0 + ac];
        v4bf b;
        b[0] = (bf16_t)v[0]; b[1] = (bf16_t)v[1];
        b[2] = (bf16_t)v[2]; b[3] = (bf16_t)v[3];
        *(v4bf*)&dB[r][ac] = b;
    }
}

__global__ __launch_bounds__(256)
void moe_up_gate_silu(const float* __restrict__ x,
                      const float* __restrict__ w_up_gate,
                      bf16_t* __restrict__ hidden) {
    __shared__ __align__(16) bf16_t lA[2][128][LSTR];
    __shared__ __align__(16) bf16_t lB[2][128][LSTR];

    const int tid  = threadIdx.x;
    const int lane = tid & 31;
    const int wav  = tid >> 5;
    const int wr   = wav >> 1;     // 0..3
    const int wc   = wav & 1;      // 0..1

    const int e  = blockIdx.z;
    const int m0 = blockIdx.x * 128;
    const int n0 = blockIdx.y * 64;

    const float* Ag = x + ((size_t)e * TPE + m0) * DDIM;
    const float* Wg = w_up_gate + (size_t)e * (2 * HDIM) * DDIM;

    v8f acc_up[2][2], acc_gt[2][2];
    #pragma unroll
    for (int i = 0; i < 2; ++i)
        #pragma unroll
        for (int j = 0; j < 2; ++j) { acc_up[i][j] = (v8f){}; acc_gt[i][j] = (v8f){}; }

    stage_k1(Ag, Wg, n0, 0, tid, lA[0], lB[0]);
    __syncthreads();

    int buf = 0;
    for (int k0 = 0; k0 < DDIM; k0 += 32) {
        if (k0 + 32 < DDIM)
            stage_k1(Ag, Wg, n0, k0 + 32, tid, lA[buf ^ 1], lB[buf ^ 1]);
        if (k0 + 64 < DDIM) {
            __builtin_prefetch(&Ag[(size_t)(tid >> 3) * DDIM + k0 + 64], 0, 0);
            __builtin_prefetch(&Wg[(size_t)(n0 + (tid >> 3)) * DDIM + k0 + 64], 0, 0);
        }

        const int l15 = lane & 15;
        const int kb  = (lane >> 4) * 8;
        const int kb2 = (lane >> 4) * 16;
        FragBF a[2], bu[2], bg[2];
        #pragma unroll
        for (int mt = 0; mt < 2; ++mt) {
            int m = wr * 32 + mt * 16 + l15;
            a[mt].h.lo = *(const v8bf*)&lA[buf][m][kb];
            a[mt].h.hi = *(const v8bf*)&lA[buf][m][kb + 16];
        }
        #pragma unroll
        for (int nt = 0; nt < 2; ++nt) {
            int n = wc * 32 + nt * 16 + l15;
            bu[nt].h.lo = *(const v8bf*)&lB[buf][n][kb];
            bu[nt].h.hi = *(const v8bf*)&lB[buf][n][kb + 16];
            bg[nt].h.lo = *(const v8bf*)&lB[buf][64 + n][kb];
            bg[nt].h.hi = *(const v8bf*)&lB[buf][64 + n][kb + 16];
        }
        #pragma unroll
        for (int mt = 0; mt < 2; ++mt)
            #pragma unroll
            for (int nt = 0; nt < 2; ++nt) {
                acc_up[mt][nt] = wmma_bf16(a[mt], bu[nt], acc_up[mt][nt]);
                acc_gt[mt][nt] = wmma_bf16(a[mt], bg[nt], acc_gt[mt][nt]);
            }
        __syncthreads();
        buf ^= 1;
    }

    const int rhalf = (lane >> 4) * 8;
    const int col   = lane & 15;
    #pragma unroll
    for (int mt = 0; mt < 2; ++mt) {
        int trow = m0 + wr * 32 + mt * 16 + rhalf;
        #pragma unroll
        for (int nt = 0; nt < 2; ++nt) {
            int h = n0 + wc * 32 + nt * 16 + col;
            bf16_t* dst = hidden + ((size_t)e * TPE + trow) * HDIM + h;
            #pragma unroll
            for (int v = 0; v < 8; ++v) {
                float u = acc_up[mt][nt][v];
                float g = acc_gt[mt][nt][v];
                dst[(size_t)v * HDIM] = (bf16_t)(u * silu_f(g));
            }
        }
    }
}

// ---------------------------------------------------------------------------
// Kernel 2: out = hidden * W_down_e. A tile (bf16, 128x32, stride H) staged by
// TDM tensor_load_to_lds with LDS padding matching LSTR; B tile transposed +
// converted manually. Block: 128 tokens x 128 d. 8 waves (4x2).
// ---------------------------------------------------------------------------
__device__ __forceinline__ void tdm_load_tileA(const bf16_t* gsrc, unsigned lds_off) {
    // 2D D# : tile 32 (dim0=h, contiguous) x 128 (dim1=t, stride HDIM elems),
    // data_size=2B. pad_interval=3 (16 DW = 64B) , pad_amount=3 (4 DW = 16B)
    // -> 80B LDS row pitch == LSTR*2.
    unsigned long long ga = (unsigned long long)(size_t)gsrc;
    u32x4 g0;
    g0[0] = 1u;                                   // count=1 valid descriptor
    g0[1] = lds_off;                              // lds_addr (bytes)
    g0[2] = (unsigned)ga;                         // global_addr[31:0]
    g0[3] = (unsigned)((ga >> 32) & 0x01FFFFFFu)  // global_addr[56:32]
          | (2u << 30);                           // type = 2 ("image")
    u32x8 g1;
    g1[0] = (1u << 16)                            // data_size = 2 bytes
          | (1u << 20)                            // pad_enable
          | (3u << 22)                            // pad_interval: 16 DWORDs
          | (3u << 25);                           // pad_amount: 4 DWORDs
    g1[1] = ((unsigned)HDIM & 0xFFFFu) << 16;     // tensor_dim0[15:0]
    g1[2] = ((unsigned)HDIM >> 16)                // tensor_dim0[31:16]
          | (((unsigned)TPE & 0xFFFFu) << 16);    // tensor_dim1[15:0]
    g1[3] = ((unsigned)TPE >> 16)
          | (32u << 16);                          // tile_dim0 = 32
    g1[4] = 128u;                                 // tile_dim1 = 128, tile_dim2 = 0
    g1[5] = (unsigned)HDIM;                       // tensor_dim0_stride (elems)
    g1[6] = 0u;
    g1[7] = 0u;
    asm volatile("tensor_load_to_lds %0, %1" :: "s"(g0), "s"(g1) : "memory");
}

__device__ __forceinline__ void stage_B_k2(const float* __restrict__ Wd,
                                           int n0, int k0, int tid,
                                           bf16_t (*dB)[LSTR]) {
    // transpose W_down[k0+h][n0+d] -> dB[d][h], fp32 -> bf16
    const int d   = tid & 127;
    const int hg0 = tid >> 7;           // 0..1
    #pragma unroll
    for (int i = 0; i < 4; ++i) {
        int h0 = (hg0 + i * 2) * 4;     // 0,8,16,24 | 4,12,20,28
        const float* src = &Wd[(size_t)(k0 + h0) * DDIM + n0 + d];
        float f0 = src[0];
        float f1 = src[DDIM];
        float f2 = src[2 * DDIM];
        float f3 = src[3 * DDIM];
        v4bf b;
        b[0] = (bf16_t)f0; b[1] = (bf16_t)f1;
        b[2] = (bf16_t)f2; b[3] = (bf16_t)f3;
        *(v4bf*)&dB[d][h0] = b;
    }
}

__global__ __launch_bounds__(256)
void moe_down(const bf16_t* __restrict__ hidden,
              const float* __restrict__ w_down,
              float* __restrict__ out) {
    __shared__ __align__(16) bf16_t lA[2][128][LSTR];
    __shared__ __align__(16) bf16_t lB[2][128][LSTR];

    const int tid  = threadIdx.x;
    const int lane = tid & 31;
    const int wav  = tid >> 5;
    const int wr   = wav >> 1;   // 0..3: 32-row group
    const int wc   = wav & 1;    // 0..1: 64-col group

    const int e  = blockIdx.z;
    const int m0 = blockIdx.x * 128;
    const int n0 = blockIdx.y * 128;

    const bf16_t* Ag = hidden + ((size_t)e * TPE + m0) * HDIM;
    const float*  Wd = w_down + (size_t)e * HDIM * DDIM;

    const unsigned ldsA0 = (unsigned)(size_t)(lds_void*)&lA[0][0][0];
    const unsigned ldsA1 = (unsigned)(size_t)(lds_void*)&lA[1][0][0];

    v8f acc[2][4];
    #pragma unroll
    for (int i = 0; i < 2; ++i)
        #pragma unroll
        for (int j = 0; j < 4; ++j) acc[i][j] = (v8f){};

    if (wav == 0) tdm_load_tileA(Ag, ldsA0);
    stage_B_k2(Wd, n0, 0, tid, lB[0]);
    if (wav == 0) __builtin_amdgcn_s_wait_tensorcnt(0);
    __syncthreads();

    int buf = 0;
    for (int k0 = 0; k0 < HDIM; k0 += 32) {
        if (k0 + 32 < HDIM) {
            if (wav == 0) tdm_load_tileA(Ag + k0 + 32, buf ? ldsA0 : ldsA1);
            stage_B_k2(Wd, n0, k0 + 32, tid, lB[buf ^ 1]);
        }

        const int l15 = lane & 15;
        const int kb  = (lane >> 4) * 8;
        const int kb2 = (lane >> 4) * 16;
        FragBF a[2], b[4];
        #pragma unroll
        for (int mt = 0; mt < 2; ++mt) {
            int m = wr * 32 + mt * 16 + l15;
            a[mt].h.lo = *(const v8bf*)&lA[buf][m][kb];
            a[mt].h.hi = *(const v8bf*)&lA[buf][m][kb + 16];
        }
        #pragma unroll
        for (int nt = 0; nt < 4; ++nt) {
            int n = wc * 64 + nt * 16 + l15;
            b[nt].h.lo = *(const v8bf*)&lB[buf][n][kb];
            b[nt].h.hi = *(const v8bf*)&lB[buf][n][kb + 16];
        }
        #pragma unroll
        for (int mt = 0; mt < 2; ++mt)
            #pragma unroll
            for (int nt = 0; nt < 4; ++nt)
                acc[mt][nt] = wmma_bf16(a[mt], b[nt], acc[mt][nt]);

        if (wav == 0) __builtin_amdgcn_s_wait_tensorcnt(0);
        __syncthreads();
        buf ^= 1;
    }

    const int rhalf = (lane >> 4) * 8;
    const int col   = lane & 15;
    #pragma unroll
    for (int mt = 0; mt < 2; ++mt) {
        int trow = m0 + wr * 32 + mt * 16 + rhalf;
        #pragma unroll
        for (int nt = 0; nt < 4; ++nt) {
            int d = n0 + wc * 64 + nt * 16 + col;
            float* dst = out + ((size_t)e * TPE + trow) * DDIM + d;
            #pragma unroll
            for (int v = 0; v < 8; ++v)
                dst[(size_t)v * DDIM] = acc[mt][nt][v];
        }
    }
}

// ---------------------------------------------------------------------------
extern "C" void kernel_launch(void* const* d_in, const int* in_sizes, int n_in,
                              void* d_out, int out_size, void* d_ws, size_t ws_size,
                              hipStream_t stream) {
    (void)in_sizes; (void)n_in; (void)out_size; (void)ws_size;
    const float* x  = (const float*)d_in[0];   // (N, D) fp32
    const float* wu = (const float*)d_in[1];   // (E, 2H, D) fp32
    const float* wd = (const float*)d_in[2];   // (E, H, D) fp32
    float*  out    = (float*)d_out;            // (N, D) fp32
    bf16_t* hidden = (bf16_t*)d_ws;            // (E, T, H) bf16 scratch (256 MB)

    dim3 g1(TPE / 128, HDIM / 64, NEXP);   // 32 x 64 x 8
    dim3 g2(TPE / 128, DDIM / 128, NEXP);  // 32 x 16 x 8
    moe_up_gate_silu<<<g1, dim3(256), 0, stream>>>(x, wu, hidden);
    moe_down<<<g2, dim3(256), 0, stream>>>(hidden, wd, out);
}
